// LULinear_48619029791063
// MI455X (gfx1250) — matrix operand order repaired and checked by reading, not compiled, not run on hardware
//
#include <hip/hip_runtime.h>
#include <hip/hip_bf16.h>

typedef __attribute__((ext_vector_type(16))) _Float16 v16h;
typedef __attribute__((ext_vector_type(8)))  float    v8f;
typedef __attribute__((ext_vector_type(2)))  float    v2f;

#define AS1 __attribute__((address_space(1)))
#define AS3 __attribute__((address_space(3)))

#if defined(__gfx1250__) && __has_builtin(__builtin_amdgcn_global_load_async_to_lds_b128) && \
    __has_builtin(__builtin_amdgcn_s_wait_asynccnt)
#define USE_ASYNC 1
#else
#define USE_ASYNC 0
#endif

#if USE_ASYNC
// Confirmed signature (from hipcc diagnostics):
//   void __builtin_amdgcn_global_load_async_to_lds_b128(i32x4* gsrc, AS3 i32x4* ldst,
//                                                       int offset, int cpol)
typedef int i32x4_vs __attribute__((vector_size(16)));
__device__ __forceinline__ void async_copy_b128(const void* g, AS3 void* l) {
  __builtin_amdgcn_global_load_async_to_lds_b128((i32x4_vs*)g, (AS3 i32x4_vs*)l, 0, 0);
}
#endif

#define B_  2048
#define D_  128
#define NT_ 64
#define R_  32
#define H_  256
#define KH_ 448    /* 3*D + NT */
#define N3_ 8192   /* 2*D*R    */

// ---------------------------------------------------------------------------
// f32 -> f16 weight conversion
// ---------------------------------------------------------------------------
__global__ void k_cvt(const float* __restrict__ src, _Float16* __restrict__ dst, int n) {
  int i = blockIdx.x * blockDim.x + threadIdx.x;
  if (i < n) dst[i] = (_Float16)src[i];
}

// ---------------------------------------------------------------------------
// h = concat(c, bmask, mmask) in f16  [B, 448]
// ---------------------------------------------------------------------------
__global__ void k_build_h(const float* __restrict__ c, const float* __restrict__ bm,
                          const float* __restrict__ mm, _Float16* __restrict__ h) {
  int i = blockIdx.x * blockDim.x + threadIdx.x;
  if (i >= B_ * KH_) return;
  int b = i / KH_, k = i % KH_;
  float v;
  if (k < 192)      v = c[b * 192 + k];
  else if (k < 320) v = bm[b * 128 + (k - 192)];
  else              v = mm[b * 128 + (k - 320)];
  h[i] = (_Float16)v;
}

// ---------------------------------------------------------------------------
// Generic f16 GEMM: C = act(A[M,K] @ Bw[K,N] + bias), WMMA f32 16x16x32 f16.
// Block: 128 threads (4 wave32). Tile 64(M) x 64(N), K-step 32.
// Staging: async global->LDS (b128) when available, else u32 via VGPRs.
// ---------------------------------------------------------------------------
template <bool OUT16, bool TANH>
__global__ void k_gemm(const _Float16* __restrict__ A, const _Float16* __restrict__ Bw,
                       const float* __restrict__ bias, void* __restrict__ Cout,
                       int M, int N, int K) {
  __shared__ _Float16 Ash[64][40];   // 64x32 + pad (row stride 80B, 16B-aligned)
  __shared__ _Float16 Bsh[32][72];   // 32x64 + pad (row stride 144B, 16B-aligned)
  const int tid  = threadIdx.x;
  const int lane = tid & 31;
  const int wid  = tid >> 5;
  const int m0 = blockIdx.x * 64;
  const int n0 = blockIdx.y * 64;

  v8f acc[4] = {};

  for (int kb = 0; kb < K; kb += 32) {
    __syncthreads();
#if USE_ASYNC
    {
      // A tile: 64 rows x 32 halves = 256 x 16B chunks (2 per thread)
      int t = tid;
      #pragma unroll
      for (int u = 0; u < 2; ++u, t += 128) {
        int r = t >> 2, c8 = (t & 3) * 8;
        async_copy_b128(&A[(size_t)(m0 + r) * K + kb + c8], (AS3 void*)&Ash[r][c8]);
      }
      // B tile: 32 rows x 64 halves = 256 x 16B chunks
      t = tid;
      #pragma unroll
      for (int u = 0; u < 2; ++u, t += 128) {
        int r = t >> 3, c8 = (t & 7) * 8;
        async_copy_b128(&Bw[(size_t)(kb + r) * N + n0 + c8], (AS3 void*)&Bsh[r][c8]);
      }
      __builtin_amdgcn_s_wait_asynccnt(0);
    }
#else
    {
      const uint32_t* Au = (const uint32_t*)A;
      const uint32_t* Bu = (const uint32_t*)Bw;
      #pragma unroll
      for (int t = tid; t < 1024; t += 128) {
        int r = t >> 4, c2 = t & 15;
        uint32_t u = Au[((size_t)(m0 + r) * K + kb) / 2 + c2];
        *(uint32_t*)&Ash[r][c2 * 2] = u;
      }
      #pragma unroll
      for (int t = tid; t < 1024; t += 128) {
        int r = t >> 5, c2 = t & 31;
        uint32_t u = Bu[((size_t)(kb + r) * N + n0) / 2 + c2];
        *(uint32_t*)&Bsh[r][c2 * 2] = u;
      }
    }
#endif
    if (kb + 32 < K) {  // prefetch next tiles -> global_prefetch_b8
      __builtin_prefetch(&A[(size_t)(m0 + (tid >> 1)) * K + kb + 32], 0, 0);
      __builtin_prefetch(&Bw[(size_t)(kb + 32 + (tid & 31)) * N + n0], 0, 0);
    }
    __syncthreads();

    // A fragment: 16x32 f16. lane L holds row M = L%16;
    // lanes<16: K 0..15; lanes>=16: K 16..31
    v16h af;
    {
      int row = wid * 16 + (lane & 15);
      int kl  = (lane & 16) ? 16 : 0;
      #pragma unroll
      for (int i = 0; i < 16; ++i) {
        int k = kl + i;
        af[i] = Ash[row][k];
      }
    }
    #pragma unroll
    for (int nt = 0; nt < 4; ++nt) {
      // B fragment: 32x16 f16. lane L holds col N = L%16;
      // lanes<16: K 0..15, lanes>=16: K 16..31
      v16h bf;
      int col = nt * 16 + (lane & 15);
      int k0  = (lane & 16) ? 16 : 0;
      #pragma unroll
      for (int i = 0; i < 16; ++i) bf[i] = Bsh[k0 + i][col];
      acc[nt] = __builtin_amdgcn_wmma_f32_16x16x32_f16(false, af, false, bf,
                                                       (short)0, acc[nt], false, false);
    }
  }

  // epilogue: C vgpr r -> (M = r + (lane>=16 ? 8:0), N = lane&15)
  int rbase = m0 + wid * 16 + ((lane & 16) ? 8 : 0);
  #pragma unroll
  for (int nt = 0; nt < 4; ++nt) {
    int col = n0 + nt * 16 + (lane & 15);
    float bv = bias[col];
    #pragma unroll
    for (int r = 0; r < 8; ++r) {
      float v = acc[nt][r] + bv;
      if (TANH) v = tanhf(v);
      size_t idx = (size_t)(rbase + r) * N + col;
      if (OUT16) ((_Float16*)Cout)[idx] = (_Float16)v;
      else       ((float*)Cout)[idx] = v;
    }
  }
}

// ---------------------------------------------------------------------------
// Per-sample fused kernel. 1 workgroup (128 threads, 4 wave32) per sample.
//  W = wc1(128x32) @ wc2(32x128) + w_eye       (f32 WMMA 16x16x4, K=32)
//  stable descending argsort of q = mmask*(1-bmask)
//  ldet = sum log|qp^2 * W[o,o] + (1-qp)|
//  y = x @ L ; z = y @ U + qp * (bc+b0)[o]     (L/U of permuted/scaled W)
// LDS ~105 KB (CDNA5 allows up to 320 KB/WG).
// ---------------------------------------------------------------------------
__global__ void k_sample(const float* __restrict__ x, const float* __restrict__ bm,
                         const float* __restrict__ mm, const float* __restrict__ weye,
                         const float* __restrict__ b0, const float* __restrict__ wc,
                         const float* __restrict__ bc, float* __restrict__ out) {
  __shared__ float Wlds[128][129];   // 66 KB
  __shared__ float wc1s[128][36];    // 18 KB (stride 144B: 16B-aligned, conflict-free frag reads)
  __shared__ float wc2s[32][132];    // 16.9 KB (stride 528B: 16B-aligned)
  __shared__ float qarr[128], xs[128], xq[128], qp[128], yq[128], red[128];
  __shared__ unsigned skey[128];
  __shared__ int sidx[128];

  const int b    = blockIdx.x;
  const int tid  = threadIdx.x;   // 0..127
  const int lane = tid & 31;
  const int wid  = tid >> 5;
  const float* wc1 = wc + (size_t)b * N3_;   // 128x32 row-major
  const float* wc2 = wc1 + D_ * R_;          // 32x128 row-major

  {
    float q = mm[b * 128 + tid] * (1.0f - bm[b * 128 + tid]);  // q >= 0
    qarr[tid] = q;
    xs[tid]   = x[b * 128 + tid];
    skey[tid] = ~__float_as_uint(q);   // ascending key == descending q, stable via idx
    sidx[tid] = tid;
  }

  // ---- stage wc1/wc2 into LDS (16B chunks; wc lives in 192MB L2) ----
#if USE_ASYNC
  for (int t = tid * 4; t < 4096; t += 512) {
    async_copy_b128(wc1 + t, (AS3 void*)&wc1s[t >> 5][t & 31]);
    async_copy_b128(wc2 + t, (AS3 void*)&wc2s[t >> 7][t & 127]);
  }
  __builtin_amdgcn_s_wait_asynccnt(0);
#else
  for (int t = tid * 4; t < 4096; t += 512) {
    *(float4*)&wc1s[t >> 5][t & 31]  = *(const float4*)(wc1 + t);
    *(float4*)&wc2s[t >> 7][t & 127] = *(const float4*)(wc2 + t);
  }
#endif
  __syncthreads();

  // ---- W = wc1 @ wc2 + w_eye via v_wmma_f32_16x16x4_f32 ----
  // 8x8 grid of 16x16 tiles; wave w computes tile-rows {2w, 2w+1}.
  // A-fragments for all 8 k-steps cached in registers, reused over 8 tile cols.
  #pragma unroll
  for (int trow = 0; trow < 2; ++trow) {
    int tm = wid * 2 + trow;
    v2f afr[8];
    {
      int ar = tm * 16 + (lane & 15);
      int ks0 = (lane & 16) ? 2 : 0;   // lanes<16: K={4k,4k+1}; lanes>=16: K={4k+2,4k+3}
      #pragma unroll
      for (int ks = 0; ks < 8; ++ks) {
        afr[ks][0] = wc1s[ar][ks * 4 + ks0];
        afr[ks][1] = wc1s[ar][ks * 4 + ks0 + 1];
      }
    }
    for (int tn = 0; tn < 8; ++tn) {
      v8f acc = {};
      int bcol = tn * 16 + (lane & 15);
      int ks0  = (lane & 16) ? 2 : 0;
      #pragma unroll
      for (int ks = 0; ks < 8; ++ks) {
        v2f bf;
        int ak = ks * 4 + ks0;
        bf[0] = wc2s[ak][bcol];
        bf[1] = wc2s[ak + 1][bcol];
        acc = __builtin_amdgcn_wmma_f32_16x16x4_f32(false, afr[ks], false, bf,
                                                    (short)0, acc, false, false);
      }
      int rr0 = tm * 16 + ((lane & 16) ? 8 : 0);
      int cc  = tn * 16 + (lane & 15);
      #pragma unroll
      for (int r = 0; r < 8; ++r) {
        int rr = rr0 + r;
        Wlds[rr][cc] = acc[r] + weye[rr * 128 + cc];
      }
    }
  }

  // ---- stable bitonic argsort (ascending on (key, idx)) ----
  for (int k = 2; k <= 128; k <<= 1) {
    for (int j = k >> 1; j >= 1; j >>= 1) {
      __syncthreads();
      int i = tid, p = tid ^ j;
      if (p > i) {
        unsigned ka = skey[i], kp = skey[p];
        int ia = sidx[i], ip = sidx[p];
        bool up = ((i & k) == 0);
        bool gt = (ka > kp) || (ka == kp && ia > ip);
        if (gt == up) { skey[i] = kp; skey[p] = ka; sidx[i] = ip; sidx[p] = ia; }
      }
    }
  }
  __syncthreads();

  qp[tid] = qarr[sidx[tid]];
  __syncthreads();
  xq[tid] = xs[tid] * qp[tid];

  // ---- log-det: Udiag_i = qp_i^2 * W[o_i,o_i] + (1 - qp_i) ----
  {
    int o = sidx[tid];
    float q = qp[tid];
    float di = q * q * Wlds[o][o] + (1.0f - q);
    red[tid] = logf(fabsf(di));
  }
  __syncthreads();
  for (int s = 64; s > 0; s >>= 1) {
    if (tid < s) red[tid] += red[tid + s];
    __syncthreads();
  }
  if (tid == 0) out[(size_t)B_ * D_ + b] = red[0];

  // ---- y = x @ L  (unit lower: y_j = x_j + qp_j * sum_{i>j} xq_i W[o_i,o_j]) ----
  {
    int oj = sidx[tid];
    float a = 0.0f;
    for (int i = tid + 1; i < 128; ++i) a += xq[i] * Wlds[sidx[i]][oj];
    float y = xs[tid] + qp[tid] * a;
    yq[tid] = y * qp[tid];
  }
  __syncthreads();

  // ---- z_k = qp_k * ( sum_{j<=k} yq_j W[o_j,o_k] + (bc+b0)[o_k] ) ----
  {
    int ok = sidx[tid];
    float a = 0.0f;
    for (int j = 0; j <= tid; ++j) a += yq[j] * Wlds[sidx[j]][ok];
    out[(size_t)b * 128 + tid] = qp[tid] * a + qp[tid] * (bc[b * 128 + ok] + b0[ok]);
  }
}

// ---------------------------------------------------------------------------
extern "C" void kernel_launch(void* const* d_in, const int* in_sizes, int n_in,
                              void* d_out, int out_size, void* d_ws, size_t ws_size,
                              hipStream_t stream) {
  (void)in_sizes; (void)n_in; (void)out_size; (void)ws_size;
  const float* x    = (const float*)d_in[0];
  const float* c    = (const float*)d_in[1];
  const float* bm   = (const float*)d_in[2];
  const float* mm   = (const float*)d_in[3];
  const float* weye = (const float*)d_in[4];
  const float* b0   = (const float*)d_in[5];
  const float* ww1  = (const float*)d_in[6];
  const float* wb1  = (const float*)d_in[7];
  const float* ww2  = (const float*)d_in[8];
  const float* wb2  = (const float*)d_in[9];
  const float* ww3  = (const float*)d_in[10];
  const float* wb3  = (const float*)d_in[11];
  const float* vw1  = (const float*)d_in[12];
  const float* vb1  = (const float*)d_in[13];
  const float* vw2  = (const float*)d_in[14];
  const float* vb2  = (const float*)d_in[15];
  const float* vw3  = (const float*)d_in[16];
  const float* vb3  = (const float*)d_in[17];

  char* ws = (char*)d_ws;
  size_t off = 0;
  auto alloc = [&](size_t bytes) { size_t o = off; off += (bytes + 255) & ~(size_t)255; return o; };

  _Float16* h16  = (_Float16*)(ws + alloc((size_t)B_ * KH_ * 2));
  _Float16* ww1h = (_Float16*)(ws + alloc((size_t)KH_ * H_ * 2));
  _Float16* vw1h = (_Float16*)(ws + alloc((size_t)KH_ * H_ * 2));
  _Float16* ww2h = (_Float16*)(ws + alloc((size_t)H_ * H_ * 2));
  _Float16* vw2h = (_Float16*)(ws + alloc((size_t)H_ * H_ * 2));
  _Float16* ww3h = (_Float16*)(ws + alloc((size_t)H_ * N3_ * 2));
  _Float16* vw3h = (_Float16*)(ws + alloc((size_t)H_ * D_ * 2));
  _Float16* t1   = (_Float16*)(ws + alloc((size_t)B_ * H_ * 2));
  _Float16* s1   = (_Float16*)(ws + alloc((size_t)B_ * H_ * 2));
  _Float16* hw   = (_Float16*)(ws + alloc((size_t)B_ * H_ * 2));
  _Float16* hb   = (_Float16*)(ws + alloc((size_t)B_ * H_ * 2));
  float*    wcb  = (float*)   (ws + alloc((size_t)B_ * N3_ * 4));   // 67 MB -> fits L2
  float*    bcb  = (float*)   (ws + alloc((size_t)B_ * D_ * 4));

  auto cvt = [&](const float* s, _Float16* d, int n) {
    k_cvt<<<(n + 255) / 256, 256, 0, stream>>>(s, d, n);
  };
  cvt(ww1, ww1h, KH_ * H_);
  cvt(vw1, vw1h, KH_ * H_);
  cvt(ww2, ww2h, H_ * H_);
  cvt(vw2, vw2h, H_ * H_);
  cvt(ww3, ww3h, H_ * N3_);
  cvt(vw3, vw3h, H_ * D_);

  k_build_h<<<(B_ * KH_ + 255) / 256, 256, 0, stream>>>(c, bm, mm, h16);

  // hypernet chains (f16 out + tanh)
  k_gemm<true, true><<<dim3(B_ / 64, H_ / 64), 128, 0, stream>>>(h16, ww1h, wb1, t1, B_, H_, KH_);
  k_gemm<true, true><<<dim3(B_ / 64, H_ / 64), 128, 0, stream>>>(h16, vw1h, vb1, s1, B_, H_, KH_);
  k_gemm<true, true><<<dim3(B_ / 64, H_ / 64), 128, 0, stream>>>(t1, ww2h, wb2, hw, B_, H_, H_);
  k_gemm<true, true><<<dim3(B_ / 64, H_ / 64), 128, 0, stream>>>(s1, vw2h, vb2, hb, B_, H_, H_);
  // wc = hw @ ww3 + wb3 (f32 out), bc = hb @ vw3 + vb3 (f32 out)
  k_gemm<false, false><<<dim3(B_ / 64, N3_ / 64), 128, 0, stream>>>(hw, ww3h, wb3, wcb, B_, N3_, H_);
  k_gemm<false, false><<<dim3(B_ / 64, D_ / 64), 128, 0, stream>>>(hb, vw3h, vb3, bcb, B_, D_, H_);

  // fused per-sample: W build + argsort + ldet + triangular solves
  k_sample<<<B_, 128, 0, stream>>>(x, bm, mm, weye, b0, wcb, bcb, (float*)d_out);
}
